// MoELayer_26525718020265
// MI455X (gfx1250) — compile-verified, matrix operand order checked by reading, not run on hardware
//
#include <hip/hip_runtime.h>
#include <hip/hip_bf16.h>
#include <math.h>

typedef __attribute__((ext_vector_type(16))) __bf16 v16bf;
typedef __attribute__((ext_vector_type(8)))  float  v8f;

#define T_TOK  8192
#define DDIM   1024
#define FDIM   4096
#define NEXP   8
#define FCHUNK 2048
#define NCHUNK (FDIM / FCHUNK)
#define BM 128
#define BN 256
#define BK 32
#define LSTR 48                         // LDS row stride (ushorts): 96B rows, 32B aligned
#define A_TILE_US   (BM * LSTR)         // 6144 ushorts
#define B_TILE_US   (BN * LSTR)         // 12288 ushorts
#define A_TILE_BYTES (A_TILE_US * 2)    // 12288 B
#define B_TILE_BYTES (B_TILE_US * 2)    // 24576 B
#define SMEM_BYTES (2 * A_TILE_BYTES + 2 * B_TILE_BYTES)   // 73728 B dynamic LDS

__device__ __forceinline__ unsigned short f2bf(float f) {
  unsigned u = __float_as_uint(f);
  unsigned r = u + 0x7FFFu + ((u >> 16) & 1u);   // round-to-nearest-even-ish
  return (unsigned short)(r >> 16);
}

// CDNA5 async global->LDS copy (ASYNCcnt-tracked, bypasses VGPRs).
__device__ __forceinline__ void async_b128(unsigned ldsoff, const void* g) {
  asm volatile("global_load_async_to_lds_b128 %0, %1, off"
               :: "v"(ldsoff), "v"(g) : "memory");
}

// ---------------------------------------------------------------- zero
__global__ __launch_bounds__(256) void moe_zero(float4* __restrict__ out4,
                                                int* __restrict__ counts) {
  size_t i = (size_t)blockIdx.x * blockDim.x + threadIdx.x;
  if (i < (size_t)T_TOK * DDIM / 4) out4[i] = make_float4(0.f, 0.f, 0.f, 0.f);
  if (blockIdx.x == 0 && threadIdx.x < NEXP) counts[threadIdx.x] = 0;
}

// ---------------------------------------------------------------- prep: f32 -> bf16
__global__ __launch_bounds__(256) void cvt_bf16(const float4* __restrict__ src,
                                                uint2* __restrict__ dst, int n4) {
  int i = blockIdx.x * 256 + threadIdx.x;
  if (i < n4) {
    float4 v = src[i];
    uint2 o;
    o.x = f2bf(v.x) | ((unsigned)f2bf(v.y) << 16);
    o.y = f2bf(v.z) | ((unsigned)f2bf(v.w) << 16);
    dst[i] = o;
  }
}

// ---------------------------------------------------------------- prep: transpose+convert
// src: z-slices of RxC fp32 (C = fast dim); dst: z-slices of CxR bf16
__global__ __launch_bounds__(256) void transpose_cvt(const float* __restrict__ src,
                                                     unsigned short* __restrict__ dst,
                                                     int R, int C) {
  __shared__ float tile[32][33];
  const size_t base = (size_t)blockIdx.z * R * C;
  const int c0 = blockIdx.x * 32;
  const int r0 = blockIdx.y * 32;
  const int tx = threadIdx.x & 31;
  const int ty = threadIdx.x >> 5;   // 0..7
#pragma unroll
  for (int i = 0; i < 32; i += 8)
    tile[ty + i][tx] = src[base + (size_t)(r0 + ty + i) * C + c0 + tx];
  __syncthreads();
#pragma unroll
  for (int i = 0; i < 32; i += 8)
    dst[base + (size_t)(c0 + ty + i) * R + r0 + tx] = f2bf(tile[tx][ty + i]);
}

// ---------------------------------------------------------------- gating
__global__ __launch_bounds__(256) void moe_gate(
    const float* __restrict__ x, const float* __restrict__ Wg,
    const float* __restrict__ bg, int* __restrict__ counts,
    int* __restrict__ idx, float* __restrict__ gates) {
  const int lane = threadIdx.x & 31;
  const int wave = threadIdx.x >> 5;
  const int t = blockIdx.x * 8 + wave;

  float a[NEXP];
#pragma unroll
  for (int e = 0; e < NEXP; ++e) a[e] = 0.f;
  const float* __restrict__ xr = x + (size_t)t * DDIM;
  for (int d = lane; d < DDIM; d += 32) {
    float xv = xr[d];
    const float4* wp = (const float4*)(Wg + (size_t)d * NEXP);
    float4 w0 = wp[0], w1 = wp[1];
    a[0] += xv * w0.x; a[1] += xv * w0.y; a[2] += xv * w0.z; a[3] += xv * w0.w;
    a[4] += xv * w1.x; a[5] += xv * w1.y; a[6] += xv * w1.z; a[7] += xv * w1.w;
  }
#pragma unroll
  for (int e = 0; e < NEXP; ++e)
#pragma unroll
    for (int off = 16; off > 0; off >>= 1) a[e] += __shfl_down(a[e], off, 32);

  if (lane == 0) {
    float lg[NEXP];
#pragma unroll
    for (int e = 0; e < NEXP; ++e) lg[e] = a[e] + bg[e];
    int i1 = 0;
#pragma unroll
    for (int e = 1; e < NEXP; ++e) if (lg[e] > lg[i1]) i1 = e;
    int i2 = (i1 == 0) ? 1 : 0;
#pragma unroll
    for (int e = 0; e < NEXP; ++e) if (e != i1 && lg[e] > lg[i2]) i2 = e;
    float p2 = expf(lg[i2] - lg[i1]);
    float s  = 1.0f + p2;
    float g[NEXP];
#pragma unroll
    for (int e = 0; e < NEXP; ++e) g[e] = 0.f;
    g[i1] = 1.0f / s;
    g[i2] = p2 / s;
#pragma unroll
    for (int e = 0; e < NEXP; ++e) gates[(size_t)t * NEXP + e] = g[e];
    int p = atomicAdd(&counts[i1], 1); idx[i1 * T_TOK + p] = t;
    p     = atomicAdd(&counts[i2], 1); idx[i2 * T_TOK + p] = t;
  }
}

// ---------------------------------------------------------------- GEMM1
// h[pos, f] = gelu( Xb[idx[pos]] @ W1t[e] + b1 )
// 128x256 block, 8 waves (2x4), 64x64 per wave -> 16 wmma / wave / k-step
__global__ __launch_bounds__(256) void moe_gemm1(
    const unsigned short* __restrict__ xb, const unsigned short* __restrict__ W1t,
    const float* __restrict__ b1, const int* __restrict__ counts,
    const int* __restrict__ idx, unsigned short* __restrict__ h,
    int e, int chunk) {
  extern __shared__ __align__(32) unsigned short smem[];
  const int count  = counts[e];
  const int tile_m = blockIdx.x / (FCHUNK / BN);
  const int tile_n = blockIdx.x % (FCHUNK / BN);
  if (tile_m * BM >= count) return;

  const int tid  = threadIdx.x;
  const int lane = tid & 31, wave = tid >> 5;
  const int wm = wave >> 2, wn = wave & 3;     // 2x4 wave grid, 64x64 each
  const int half = lane >> 4, l16 = lane & 15;

  // A staging: 2 threads/row, each 2x16B chunks
  const int srow = tid >> 1;                   // 0..127
  const int aseg = (tid & 1) * 16;             // bf16 elems
  const int arow_g = tile_m * BM + srow;
  const int tok = (arow_g < count) ? idx[e * T_TOK + arow_g] : 0;
  const unsigned short* __restrict__ ag = xb + (size_t)tok * DDIM + aseg;
  // B staging: 1 thread/row, 4x16B chunks
  const int fbase = chunk * FCHUNK + tile_n * BN;
  const unsigned short* __restrict__ bgp =
      W1t + ((size_t)e * FDIM + fbase + tid) * DDIM;           // W1t[e][f][d]
  const unsigned lds_base = (unsigned)(uintptr_t)smem;
  const unsigned aoff = (unsigned)(srow * 96 + aseg * 2);
  const unsigned boff = (unsigned)(tid * 96);

  v8f zero8 = {0.f, 0.f, 0.f, 0.f, 0.f, 0.f, 0.f, 0.f};
  v8f acc[4][4];
#pragma unroll
  for (int mt = 0; mt < 4; ++mt)
#pragma unroll
    for (int nt = 0; nt < 4; ++nt) acc[mt][nt] = zero8;

  auto issue = [&](int kk, int buf) {     // 6 async instructions per wave
    unsigned ab = lds_base + (unsigned)buf * A_TILE_BYTES + aoff;
    async_b128(ab,       ag + kk);
    async_b128(ab + 16u, ag + kk + 8);
    unsigned bb = lds_base + 2u * A_TILE_BYTES + (unsigned)buf * B_TILE_BYTES + boff;
#pragma unroll
    for (int c = 0; c < 4; ++c)
      async_b128(bb + 16u * c, bgp + kk + 8 * c);
  };
  // low-pressure compute: bfr[4] resident, single A fragment streamed per mt
  auto compute = [&](int buf) {
    const unsigned short* Ab = smem + buf * A_TILE_US;
    const unsigned short* Bb = smem + 2 * A_TILE_US + buf * B_TILE_US;
    v16bf bfr[4];
#pragma unroll
    for (int nt = 0; nt < 4; ++nt)
      bfr[nt] = *(const v16bf*)&Bb[(wn * 64 + nt * 16 + l16) * LSTR + half * 16];
#pragma unroll
    for (int mt = 0; mt < 4; ++mt) {
      v16bf a = *(const v16bf*)&Ab[(wm * 64 + mt * 16 + l16) * LSTR + half * 16];
#pragma unroll
      for (int nt = 0; nt < 4; ++nt)
        acc[mt][nt] = __builtin_amdgcn_wmma_f32_16x16x32_bf16(
            false, a, false, bfr[nt], (short)0, acc[mt][nt], false, false);
    }
  };

  issue(0, 0);
  int buf = 0;
#pragma unroll 1
  for (int k0 = 0; k0 < DDIM - BK; k0 += BK) {
    issue(k0 + BK, buf ^ 1);
    asm volatile("s_wait_asynccnt 0x6" ::: "memory");   // prev batch landed
    __syncthreads();
    compute(buf);
    __syncthreads();
    buf ^= 1;
  }
  asm volatile("s_wait_asynccnt 0x0" ::: "memory");
  __syncthreads();
  compute(buf);

  // epilogue: bias + exact GeLU -> bf16 h
#pragma unroll
  for (int mt = 0; mt < 4; ++mt) {
#pragma unroll
    for (int nt = 0; nt < 4; ++nt) {
      const int col = tile_n * BN + wn * 64 + nt * 16 + l16;   // [0,FCHUNK)
      const float bias = b1[e * FDIM + chunk * FCHUNK + col];
#pragma unroll
      for (int r = 0; r < 8; ++r) {
        const int row_g = tile_m * BM + wm * 64 + mt * 16 + half * 8 + r;
        if (row_g < count) {
          float v = acc[mt][nt][r] + bias;
          v = 0.5f * v * (1.0f + erff(v * 0.70710678118654752f));
          h[(size_t)row_g * FCHUNK + col] = f2bf(v);
        }
      }
    }
  }
}

// ---------------------------------------------------------------- GEMM2
// out[tok] += gate * (h @ W2t[e] + (chunk==0 ? b2 : 0))
__global__ __launch_bounds__(256) void moe_gemm2(
    const unsigned short* __restrict__ h, const unsigned short* __restrict__ W2t,
    const float* __restrict__ b2, const int* __restrict__ counts,
    const int* __restrict__ idx, const float* __restrict__ gates,
    float* __restrict__ out, int e, int chunk) {
  extern __shared__ __align__(32) unsigned short smem[];
  const int count  = counts[e];
  const int tile_m = blockIdx.x / (DDIM / BN);
  const int tile_n = blockIdx.x % (DDIM / BN);
  if (tile_m * BM >= count) return;

  const int tid  = threadIdx.x;
  const int lane = tid & 31, wave = tid >> 5;
  const int wm = wave >> 2, wn = wave & 3;
  const int half = lane >> 4, l16 = lane & 15;

  const int srow = tid >> 1;
  const int aseg = (tid & 1) * 16;
  const unsigned short* __restrict__ ag =
      h + (size_t)(tile_m * BM + srow) * FCHUNK + aseg;
  const unsigned short* __restrict__ bgp =
      W2t + ((size_t)e * DDIM + tile_n * BN + tid) * FDIM + chunk * FCHUNK;
  const unsigned lds_base = (unsigned)(uintptr_t)smem;
  const unsigned aoff = (unsigned)(srow * 96 + aseg * 2);
  const unsigned boff = (unsigned)(tid * 96);

  v8f zero8 = {0.f, 0.f, 0.f, 0.f, 0.f, 0.f, 0.f, 0.f};
  v8f acc[4][4];
#pragma unroll
  for (int mt = 0; mt < 4; ++mt)
#pragma unroll
    for (int nt = 0; nt < 4; ++nt) acc[mt][nt] = zero8;

  auto issue = [&](int kk, int buf) {
    unsigned ab = lds_base + (unsigned)buf * A_TILE_BYTES + aoff;
    async_b128(ab,       ag + kk);
    async_b128(ab + 16u, ag + kk + 8);
    unsigned bb = lds_base + 2u * A_TILE_BYTES + (unsigned)buf * B_TILE_BYTES + boff;
#pragma unroll
    for (int c = 0; c < 4; ++c)
      async_b128(bb + 16u * c, bgp + kk + 8 * c);
  };
  auto compute = [&](int buf) {
    const unsigned short* Ab = smem + buf * A_TILE_US;
    const unsigned short* Bb = smem + 2 * A_TILE_US + buf * B_TILE_US;
    v16bf bfr[4];
#pragma unroll
    for (int nt = 0; nt < 4; ++nt)
      bfr[nt] = *(const v16bf*)&Bb[(wn * 64 + nt * 16 + l16) * LSTR + half * 16];
#pragma unroll
    for (int mt = 0; mt < 4; ++mt) {
      v16bf a = *(const v16bf*)&Ab[(wm * 64 + mt * 16 + l16) * LSTR + half * 16];
#pragma unroll
      for (int nt = 0; nt < 4; ++nt)
        acc[mt][nt] = __builtin_amdgcn_wmma_f32_16x16x32_bf16(
            false, a, false, bfr[nt], (short)0, acc[mt][nt], false, false);
    }
  };

  issue(0, 0);
  int buf = 0;
#pragma unroll 1
  for (int k0 = 0; k0 < FCHUNK - BK; k0 += BK) {
    issue(k0 + BK, buf ^ 1);
    asm volatile("s_wait_asynccnt 0x6" ::: "memory");
    __syncthreads();
    compute(buf);
    __syncthreads();
    buf ^= 1;
  }
  asm volatile("s_wait_asynccnt 0x0" ::: "memory");
  __syncthreads();
  compute(buf);

  // epilogue: scatter-accumulate with gate weight (launches serialize -> no race)
#pragma unroll
  for (int mt = 0; mt < 4; ++mt) {
#pragma unroll
    for (int r = 0; r < 8; ++r) {
      const int row_g = tile_m * BM + wm * 64 + mt * 16 + half * 8 + r;
      if (row_g < count) {
        const int tok = idx[e * T_TOK + row_g];
        const float g = gates[(size_t)tok * NEXP + e];
#pragma unroll
        for (int nt = 0; nt < 4; ++nt) {
          const int col = tile_n * BN + wn * 64 + nt * 16 + l16;
          float v = acc[mt][nt][r];
          if (chunk == 0) v += b2[e * DDIM + col];
          out[(size_t)tok * DDIM + col] += g * v;
        }
      }
    }
  }
}

// ---------------------------------------------------------------- launch
extern "C" void kernel_launch(void* const* d_in, const int* in_sizes, int n_in,
                              void* d_out, int out_size, void* d_ws, size_t ws_size,
                              hipStream_t stream) {
  const float* x  = (const float*)d_in[0];
  const float* W1 = (const float*)d_in[1];
  const float* b1 = (const float*)d_in[2];
  const float* W2 = (const float*)d_in[3];
  const float* b2 = (const float*)d_in[4];
  const float* Wg = (const float*)d_in[5];
  const float* bg = (const float*)d_in[6];
  // d_in[7] = top_k (== 2, hardcoded in routing)
  float* out = (float*)d_out;

  // workspace layout (~177 MB):
  //   counts[E] | idx[E][T] | gates[T][E] | xb bf16 | W1t bf16 | W2t bf16 | h bf16
  char* ws = (char*)d_ws;
  size_t off = 0;
  int* counts = (int*)(ws + off);                 off += 256;
  int* idx    = (int*)(ws + off);                 off += (size_t)NEXP * T_TOK * 4;
  float* gates = (float*)(ws + off);              off += (size_t)T_TOK * NEXP * 4;
  unsigned short* xb  = (unsigned short*)(ws + off); off += (size_t)T_TOK * DDIM * 2;
  unsigned short* W1t = (unsigned short*)(ws + off); off += (size_t)NEXP * DDIM * FDIM * 2;
  unsigned short* W2t = (unsigned short*)(ws + off); off += (size_t)NEXP * FDIM * DDIM * 2;
  unsigned short* h   = (unsigned short*)(ws + off);

  moe_zero<<<(T_TOK * DDIM / 4 + 255) / 256, 256, 0, stream>>>((float4*)out, counts);
  cvt_bf16<<<(T_TOK * DDIM / 4 + 255) / 256, 256, 0, stream>>>(
      (const float4*)x, (uint2*)xb, T_TOK * DDIM / 4);
  transpose_cvt<<<dim3(FDIM / 32, DDIM / 32, NEXP), 256, 0, stream>>>(W1, W1t, DDIM, FDIM);
  transpose_cvt<<<dim3(DDIM / 32, FDIM / 32, NEXP), 256, 0, stream>>>(W2, W2t, FDIM, DDIM);
  moe_gate<<<T_TOK / 8, 256, 0, stream>>>(x, Wg, bg, counts, idx, gates);

  for (int e = 0; e < NEXP; ++e) {
    for (int c = 0; c < NCHUNK; ++c) {
      moe_gemm1<<<(T_TOK / BM) * (FCHUNK / BN), 256, SMEM_BYTES, stream>>>(
          xb, W1t, b1, counts, idx, h, e, c);
      moe_gemm2<<<(T_TOK / BM) * (DDIM / BN), 256, SMEM_BYTES, stream>>>(
          h, W2t, b2, counts, idx, gates, out, e, c);
    }
  }
}
